// DeepSetPred_35828617183719
// MI455X (gfx1250) — matrix operand order repaired and checked by reading, not compiled, not run on hardware
//
#include <hip/hip_runtime.h>
#include <hip/hip_bf16.h>
#include <math.h>

// ---------------------------------------------------------------------------
// Types for CDNA5 WMMA (wave32): v_wmma_f32_16x16x32_bf16
// ---------------------------------------------------------------------------
typedef __bf16 bf16_t;
typedef __attribute__((ext_vector_type(8)))  bf16_t bf16x8;
typedef __attribute__((ext_vector_type(16))) bf16_t v16bf;
typedef __attribute__((ext_vector_type(8)))  float  v8f;

__device__ __forceinline__ bf16_t to_bf16(float f) { return (bf16_t)f; }

// Hardware transcendental tanh on gfx1250 (V_TANH_F32) -- one TRANS32 op
// instead of OCML's ~30-op branchy exp-based expansion. TRANS32 ops
// co-execute with the WMMA (XDL) pipe, so this removes the VALU bottleneck.
__device__ __forceinline__ float fast_tanh(float x) {
#if __has_builtin(__builtin_amdgcn_tanhf)
    return __builtin_amdgcn_tanhf(x);
#elif __has_builtin(__builtin_amdgcn_tanh_f32)
    return __builtin_amdgcn_tanh_f32(x);
#else
    return tanhf(x);
#endif
}

union AB16 {
    v16bf v;
    struct { bf16x8 lo, hi; } h;
};

// ---------------------------------------------------------------------------
// Weight packing: W [K,N] row-major f32 -> bf16 in WMMA B-fragment order.
// Tile (nt,kt) = 32 lanes x 16 bf16, contiguous. Lane n<16 holds column
// nt*16+n with K = kt*32 + 0..15; lane n+16 holds same column, K = 16..31.
// Each lane's 16 values are contiguous -> single 32B load in the GEMM.
// ---------------------------------------------------------------------------
__global__ void pack_weight_kernel(const float* __restrict__ src,
                                   bf16_t* __restrict__ dst, int K, int N)
{
    int idx = blockIdx.x * blockDim.x + threadIdx.x;
    int total = K * N;
    if (idx >= total) return;
    int numKt   = K >> 5;
    int tile    = idx >> 9;          // 512 elems per tile
    int within  = idx & 511;
    int lane    = within >> 4;
    int i       = within & 15;
    int nt      = tile / numKt;
    int kt      = tile - nt * numKt;
    int n       = nt * 16 + (lane & 15);
    int k       = kt * 32 + ((lane >> 4) << 4) + i;
    dst[idx] = to_bf16(src[k * N + n]);
}

__global__ void zero_kernel(float* __restrict__ p, int n)
{
    int i = blockIdx.x * blockDim.x + threadIdx.x;
    if (i < n) p[i] = 0.f;
}

// ---------------------------------------------------------------------------
// One fused MLP layer on a 32-row token tile held in LDS.
//   out[32][N] = act( A[32][K](bf16,LDS) @ Wp(packed bf16,global/L2) + bias )
// 8 waves: wave&1 -> M-tile (rows 0-15 / 16-31), wave>>1 strides N-tiles.
// A-fragment (contiguous K, same order as packed B): lane L<16 reads row mt*16+L,
// K 0..15; lane L+16 same row, K 16..31.
// ---------------------------------------------------------------------------
template<int K, int N, bool TANH>
__device__ __forceinline__ void mlp_layer(const bf16_t* __restrict__ A,
                                          const bf16_t* __restrict__ Wp,
                                          const float* __restrict__ bias,
                                          bf16_t* __restrict__ outb,
                                          float*  __restrict__ outf,
                                          int wave, int lane)
{
    constexpr int numKt = K / 32;
    constexpr int numNt = N / 16;
    const int mt     = wave & 1;
    const int laneN  = lane & 15;
    const int laneHi = lane >> 4;            // 0 or 1
    const int rowA   = mt * 16 + laneN;
    const int kbase  = laneHi * 16;          // first K chunk offset

    for (int nt = (wave >> 1); nt < numNt; nt += 4) {
        v8f acc = {};
        const bf16_t* wptr = Wp + (size_t)(nt * numKt) * 512 + lane * 16;
        #pragma unroll
        for (int kt = 0; kt < numKt; ++kt) {
            AB16 a;
            const bf16_t* ap = A + rowA * K + kt * 32 + kbase;
            a.h.lo = *(const bf16x8*)(ap);        // 16B ds_load
            a.h.hi = *(const bf16x8*)(ap + 8);    // 16B ds_load
            v16bf b = *(const v16bf*)(wptr);      // 32B coalesced global load
            wptr += 512;
            acc = __builtin_amdgcn_wmma_f32_16x16x32_bf16(
                      false, a.v, false, b, (short)0, acc, false, false);
        }
        const int n  = nt * 16 + laneN;
        const float bv = bias[n];
        #pragma unroll
        for (int r = 0; r < 8; ++r) {
            // C/D f32 layout: VGPR r -> M = mt*16 + laneHi*8 + r, N = laneN
            const int m = mt * 16 + laneHi * 8 + r;
            float v = acc[r] + bv;
            if constexpr (TANH) outb[m * N + n] = to_bf16(fast_tanh(v));
            else                outf[m * N + n] = v;
        }
    }
}

// ---------------------------------------------------------------------------
// Fused encoder: words tile -> 3-layer MLP (WMMA bf16) -> per-block segment
// pre-reduction -> global f32 atomics into enc[128][256].
// LDS (64KB, overlaid):
//   [0,32K):  words bf16 [32][256] (16K), later h2 bf16 [32][512] (32K)
//   [32K,64K): h1 bf16 [32][512] (32K),   later tok_enc f32 [32][256] (32K)
// ---------------------------------------------------------------------------
#define ROWS 32

__global__ __launch_bounds__(256) void encoder_kernel(
    const float* __restrict__ words, const int* __restrict__ seg_ids,
    const bf16_t* __restrict__ W1p, const float* __restrict__ b1,
    const bf16_t* __restrict__ W2p, const float* __restrict__ b2,
    const bf16_t* __restrict__ W3p, const float* __restrict__ b3,
    float* __restrict__ enc)
{
    __shared__ __align__(32) char smem[65536];
    __shared__ int sseg[ROWS];
    bf16_t* regionA = (bf16_t*)smem;             // words, then h2
    bf16_t* h1      = (bf16_t*)(smem + 32768);   // h1
    float*  tokenc  = (float*)(smem + 32768);    // tok_enc overlays h1

    const int tid = threadIdx.x;
    const int t0  = blockIdx.x * ROWS;

    // Stage + convert words tile (coalesced), fetch segment ids.
    for (int e = tid; e < ROWS * 256; e += 256)
        regionA[e] = to_bf16(words[(size_t)t0 * 256 + e]);
    if (tid < ROWS) sseg[tid] = seg_ids[t0 + tid];
    __syncthreads();

    const int wave = tid >> 5, lane = tid & 31;

    // Layer 1: words[32][256] -> h1[32][512], tanh
    mlp_layer<256, 512, true >(regionA, W1p, b1, h1, nullptr, wave, lane);
    __syncthreads();
    // Layer 2: h1[32][512] -> h2[32][512] (over words region), tanh
    mlp_layer<512, 512, true >(h1, W2p, b2, regionA, nullptr, wave, lane);
    __syncthreads();
    // Layer 3: h2[32][512] -> tok_enc f32 [32][256] (over h1 region)
    mlp_layer<512, 256, false>(regionA, W3p, b3, nullptr, tokenc, wave, lane);
    __syncthreads();

    // Segment pre-reduction: one thread per output column; seg_ids sorted so
    // a 32-row block emits ~1 atomic per column instead of 32.
    {
        const int c = tid;            // 0..255
        float acc = 0.f;
        int cur = sseg[0];
        for (int m = 0; m < ROWS; ++m) {
            int s = sseg[m];
            if (s != cur) {
                atomicAdd(&enc[cur * 256 + c], acc);
                acc = 0.f;
                cur = s;
            }
            acc += tokenc[m * 256 + c];
        }
        atomicAdd(&enc[cur * 256 + c], acc);
    }
}

// ---------------------------------------------------------------------------
// Predictor: tiny (128 rows) -> plain VALU dense layers.
// ---------------------------------------------------------------------------
template<int IN, int OUT, bool TANH>
__global__ void dense_kernel(const float* __restrict__ x,
                             const float* __restrict__ W,
                             const float* __restrict__ b,
                             float* __restrict__ y, int rows)
{
    int idx = blockIdx.x * blockDim.x + threadIdx.x;
    if (idx >= rows * OUT) return;
    int i = idx / OUT, j = idx % OUT;
    float acc = b[j];
    const float* xr = x + i * IN;
    #pragma unroll 4
    for (int k = 0; k < IN; ++k)
        acc = fmaf(xr[k], W[k * OUT + j], acc);
    y[idx] = TANH ? fast_tanh(acc) : acc;
}

// ---------------------------------------------------------------------------
extern "C" void kernel_launch(void* const* d_in, const int* in_sizes, int n_in,
                              void* d_out, int out_size, void* d_ws, size_t ws_size,
                              hipStream_t stream)
{
    const float* words  = (const float*)d_in[0];
    const int*   seg    = (const int*)  d_in[1];
    const float* W1 = (const float*)d_in[2];  const float* b1  = (const float*)d_in[3];
    const float* W2 = (const float*)d_in[4];  const float* b2  = (const float*)d_in[5];
    const float* W3 = (const float*)d_in[6];  const float* b3  = (const float*)d_in[7];
    const float* P1 = (const float*)d_in[8];  const float* pb1 = (const float*)d_in[9];
    const float* P2 = (const float*)d_in[10]; const float* pb2 = (const float*)d_in[11];
    const float* P3 = (const float*)d_in[12]; const float* pb3 = (const float*)d_in[13];

    char* ws = (char*)d_ws;
    bf16_t* W1p = (bf16_t*)(ws + 0);         // 256*512 bf16 = 256KB
    bf16_t* W2p = (bf16_t*)(ws + 262144);    // 512*512 bf16 = 512KB
    bf16_t* W3p = (bf16_t*)(ws + 786432);    // 512*256 bf16 = 256KB
    float*  enc = (float*)(ws + 1048576);    // 128*256 f32  = 128KB
    float*  p1  = (float*)(ws + 1179648);    // 128*512 f32  = 256KB
    float*  p2  = (float*)(ws + 1441792);    // 128*512 f32  = 256KB

    // Pack weights into WMMA B-fragment layout (bf16); zero accumulator.
    pack_weight_kernel<<<512, 256, 0, stream>>>(W1, W1p, 256, 512);
    pack_weight_kernel<<<1024, 256, 0, stream>>>(W2, W2p, 512, 512);
    pack_weight_kernel<<<512, 256, 0, stream>>>(W3, W3p, 512, 256);
    zero_kernel<<<128, 256, 0, stream>>>(enc, 128 * 256);

    // Fused WMMA encoder + segment sum: 131072 tokens / 32 rows per block.
    encoder_kernel<<<131072 / ROWS, 256, 0, stream>>>(
        words, seg, W1p, b1, W2p, b2, W3p, b3, enc);

    // Predictor MLP (128 rows, negligible FLOPs).
    dense_kernel<256, 512, true ><<<(128 * 512 + 255) / 256, 256, 0, stream>>>(enc, P1, pb1, p1, 128);
    dense_kernel<512, 512, true ><<<(128 * 512 + 255) / 256, 256, 0, stream>>>(p1,  P2, pb2, p2, 128);
    dense_kernel<512, 32,  false><<<(128 * 32  + 255) / 256, 256, 0, stream>>>(p2,  P3, pb3, (float*)d_out, 128);
}
